// MultiHeadAttention_62534723829844
// MI455X (gfx1250) — matrix operand order repaired and checked by reading, not compiled, not run on hardware
//
#include <hip/hip_runtime.h>

typedef __attribute__((ext_vector_type(16))) __bf16 v16bf;
typedef __attribute__((ext_vector_type(8)))  __bf16 v8bf;
typedef __attribute__((ext_vector_type(8)))  float  v8f;

#define B_SZ   2
#define S_LEN  2048
#define D_MOD  2048
#define N_HEAD 16
#define D_HEAD 128
#define M_ROWS (B_SZ * S_LEN)   // 4096

static __device__ inline v8f vzero8() {
    v8f z;
#pragma unroll
    for (int i = 0; i < 8; ++i) z[i] = 0.0f;
    return z;
}

static __device__ inline v8f wmma_bf16(v16bf a, v16bf b, v8f c) {
    // D = A(16x32) * B(32x16) + C(16x16), f32 accumulate
    return __builtin_amdgcn_wmma_f32_16x16x32_bf16(
        /*neg_a=*/false, a, /*neg_b=*/false, b,
        /*c_mod=*/(short)0, c, /*reuse_a=*/false, /*reuse_b=*/false);
}

// 16-byte global -> LDS async copy (ASYNCcnt-tracked, no VGPR staging)
static __device__ inline void async_copy16(uint32_t lds_addr, const void* gaddr) {
    asm volatile("global_load_async_to_lds_b128 %0, %1, off"
                 :: "v"(lds_addr), "v"(gaddr) : "memory");
}
static __device__ inline void wait_async0() {
    asm volatile("s_wait_asynccnt 0" ::: "memory");
}
// async loads complete in order: <=8 outstanding means the older batch landed
static __device__ inline void wait_async8() {
    asm volatile("s_wait_asynccnt 8" ::: "memory");
}

union bfrag { v16bf v; v8bf h2[2]; };

// ---------------------------------------------------------------------------
// fp32 -> bf16 conversion (4 elements / thread)
// ---------------------------------------------------------------------------
__global__ void cvt_f32_to_bf16(const float* __restrict__ in,
                                __bf16* __restrict__ out, int n) {
    int i = (blockIdx.x * blockDim.x + threadIdx.x) * 4;
    if (i + 3 < n) {
        float4 f = *(const float4*)(in + i);
        out[i + 0] = (__bf16)f.x;
        out[i + 1] = (__bf16)f.y;
        out[i + 2] = (__bf16)f.z;
        out[i + 3] = (__bf16)f.w;
    }
}

// ---------------------------------------------------------------------------
// C[m,n] = sum_k A[m,k] * W[n,k]   (A: MxK row-major, W: NxK row-major)
// Block: 256 threads (8 waves), tile 128(M) x 128(N), staged K-chunks of 64.
// Double-buffered async-to-LDS staging: stage s+1 is in flight while the
// 16 WMMAs of stage s execute (s_wait_asynccnt 8 = wait for the older batch).
// ---------------------------------------------------------------------------
template <bool BF16OUT>
__global__ __launch_bounds__(256) void gemm_bf16_nt(
    const __bf16* __restrict__ A, const __bf16* __restrict__ W,
    void* __restrict__ Cout, int M, int N, int K) {
    __shared__ __bf16 lA[2][128][72];   // ping-pong, padded stride
    __shared__ __bf16 lW[2][128][72];
    constexpr uint32_t BUF_BYTES = 128 * 72 * sizeof(__bf16);  // 18432

    const int tid  = threadIdx.x;
    const int lane = tid & 31;
    const int w    = tid >> 5;
    const int l15  = lane & 15;
    const int hi8  = (lane & 16) ? 8 : 0;
    const int kb16 = (lane & 16) ? 16 : 0;
    const int m0   = blockIdx.y * 128;
    const int n0   = blockIdx.x * 128;

    v8f acc[8];
#pragma unroll
    for (int i = 0; i < 8; ++i) acc[i] = vzero8();

    // cooperative staging assignment: thread -> (row, 32-col half), 4x16B each
    const int r    = tid >> 1;       // 0..127
    const int half = tid & 1;        // 0..1
    const __bf16* gA = A + (size_t)(m0 + r) * K + half * 32;
    const __bf16* gW = W + (size_t)(n0 + r) * K + half * 32;
    const uint32_t ldsA = (uint32_t)(uintptr_t)&lA[0][r][half * 32];
    const uint32_t ldsW = (uint32_t)(uintptr_t)&lW[0][r][half * 32];

    auto prefetch_stage = [&](int s) {
        const int kc = s * 64;
        const uint32_t boff = (uint32_t)(s & 1) * BUF_BYTES;
#pragma unroll
        for (int j = 0; j < 4; ++j) {
            async_copy16(ldsA + boff + j * 16, gA + kc + j * 8);
            async_copy16(ldsW + boff + j * 16, gW + kc + j * 8);
        }
    };

    const int nstage = K / 64;
    prefetch_stage(0);
    for (int s = 0; s < nstage; ++s) {
        if (s + 1 < nstage) {
            prefetch_stage(s + 1);   // safe: end-of-stage barrier of s-1 passed
            wait_async8();           // stage s landed; stage s+1 still in flight
        } else {
            wait_async0();
        }
        __syncthreads();

        const int buf = s & 1;
#pragma unroll
        for (int sub = 0; sub < 64; sub += 32) {
            bfrag a;   // A-frag: row = w*16 + l15, k = hi8+{0..7}, 16+hi8+{0..7}
            a.h2[0] = *(const v8bf*)&lA[buf][w * 16 + l15][sub + hi8];
            a.h2[1] = *(const v8bf*)&lA[buf][w * 16 + l15][sub + 16 + hi8];
#pragma unroll
            for (int nt = 0; nt < 8; ++nt) {
                bfrag b;  // B-frag: n = nt*16 + l15, k = hi8+{0..7}, 16+hi8+{0..7}
                b.h2[0] = *(const v8bf*)&lW[buf][nt * 16 + l15][sub + hi8];
                b.h2[1] = *(const v8bf*)&lW[buf][nt * 16 + l15][sub + 16 + hi8];
                acc[nt] = wmma_bf16(a.v, b.v, acc[nt]);
            }
        }
        __syncthreads();
    }

#pragma unroll
    for (int nt = 0; nt < 8; ++nt) {
#pragma unroll
        for (int rr = 0; rr < 8; ++rr) {
            int row = m0 + w * 16 + rr + hi8;     // C layout: M = r + (lane>=16 ? 8:0)
            int col = n0 + nt * 16 + l15;
            if constexpr (BF16OUT)
                ((__bf16*)Cout)[(size_t)row * N + col] = (__bf16)acc[nt][rr];
            else
                ((float*)Cout)[(size_t)row * N + col] = acc[nt][rr];
        }
    }
}

// ---------------------------------------------------------------------------
// Causal flash attention over bf16 Q/K/V laid out (B, S, H*DK).
// Block: 128 threads (4 waves) = 64 queries of one (b, h).
// Per 32-key kv-block: K tile staged via async-to-LDS, V transposed in LDS,
// scores via 8 WMMA, online softmax in f32 C-fragments, P->LDS->A-frag,
// PV via 8 WMMA.
// ---------------------------------------------------------------------------
__global__ __launch_bounds__(128) void flash_attn(
    const __bf16* __restrict__ Qg, const __bf16* __restrict__ Kg,
    const __bf16* __restrict__ Vg, __bf16* __restrict__ Og) {
    __shared__ __bf16 lK[32][136];      // 32 keys x 128 dk (row-major, padded)
    __shared__ __bf16 lVt[128][40];     // transposed: [dk][key] (padded)
    __shared__ __bf16 lP[4][16][40];    // per-wave P tile 16 x 32 (padded)

    const int tid  = threadIdx.x;
    const int lane = tid & 31;
    const int w    = tid >> 5;
    const int l15  = lane & 15;
    const int hi8  = (lane & 16) ? 8 : 0;
    const int kb16 = (lane & 16) ? 16 : 0;

    const int bh   = blockIdx.y;
    const int b    = bh / N_HEAD;
    const int h    = bh % N_HEAD;
    const int qb0  = blockIdx.x * 64;
    const int q0   = qb0 + w * 16;

    const size_t base = (size_t)b * S_LEN * D_MOD + (size_t)h * D_HEAD;
    const __bf16* Qp = Qg + base;
    const __bf16* Kp = Kg + base;
    const __bf16* Vp = Vg + base;
    __bf16*       Op = Og + base;

    // Q A-fragments, one per 32-wide dk chunk (row = q0 + l15)
    bfrag aq[4];
    {
        const __bf16* qrow = Qp + (size_t)(q0 + l15) * D_MOD;
#pragma unroll
        for (int c = 0; c < 4; ++c) {
            aq[c].h2[0] = *(const v8bf*)(qrow + c * 32 + hi8);
            aq[c].h2[1] = *(const v8bf*)(qrow + c * 32 + 16 + hi8);
        }
    }

    v8f o[8];
#pragma unroll
    for (int i = 0; i < 8; ++i) o[i] = vzero8();
    float mrow[8], lrow[8];
#pragma unroll
    for (int i = 0; i < 8; ++i) { mrow[i] = -__builtin_inff(); lrow[i] = 0.0f; }

    const float scale = 0.08838834764831845f;  // 1/sqrt(128)
    const int nkv = qb0 / 32 + 2;              // causal extent for this block

    // staging assignment for the K tile (async): thread -> (key row, 32-col quarter)
    const int krow = tid >> 2, kquad = tid & 3;
    const __bf16* ksrc0 = Kp + (size_t)krow * D_MOD + kquad * 32;
    const uint32_t kdst = (uint32_t)(uintptr_t)&lK[krow][kquad * 32];
    // staging assignment for the V tile (transpose via VGPR)
    const int vrow = tid & 31, vg = tid >> 5;
    const __bf16* vsrc0 = Vp + (size_t)vrow * D_MOD + vg * 32;

    for (int kv = 0; kv < nkv; ++kv) {
        const int k0 = kv * 32;
        { // K tile: 32 rows x 128 dk, async b128 copies (64 B / thread)
            const __bf16* ksrc = ksrc0 + (size_t)k0 * D_MOD;
#pragma unroll
            for (int j = 0; j < 4; ++j)
                async_copy16(kdst + j * 16, ksrc + j * 8);
        }
        { // V tile transposed into lVt[dk][key]
            const __bf16* vsrc = vsrc0 + (size_t)k0 * D_MOD;
#pragma unroll
            for (int j = 0; j < 4; ++j) {
                v8bf vv = *(const v8bf*)(vsrc + j * 8);
#pragma unroll
                for (int u = 0; u < 8; ++u) lVt[vg * 32 + j * 8 + u][vrow] = vv[u];
            }
        }
        wait_async0();
        __syncthreads();

        if (k0 <= q0 + 15) {   // wave has at least one unmasked key in this block
            v8f s0 = vzero8(), s1 = vzero8();
#pragma unroll
            for (int c = 0; c < 4; ++c) {
                bfrag b0, b1;  // B[k=dk][n=key] = K[key][dk]
                b0.h2[0] = *(const v8bf*)&lK[l15][c * 32 + hi8];
                b0.h2[1] = *(const v8bf*)&lK[l15][c * 32 + 16 + hi8];
                b1.h2[0] = *(const v8bf*)&lK[16 + l15][c * 32 + hi8];
                b1.h2[1] = *(const v8bf*)&lK[16 + l15][c * 32 + 16 + hi8];
                s0 = wmma_bf16(aq[c].v, b0.v, s0);
                s1 = wmma_bf16(aq[c].v, b1.v, s1);
            }

            // online softmax: row stats reduced across the 16 N-lanes
#pragma unroll
            for (int rr = 0; rr < 8; ++rr) {
                int qi = q0 + rr + hi8;
                float x0 = s0[rr] * scale; if (k0 + l15 > qi)      x0 = -__builtin_inff();
                float x1 = s1[rr] * scale; if (k0 + 16 + l15 > qi) x1 = -__builtin_inff();
                float mx = fmaxf(x0, x1);
                mx = fmaxf(mx, __shfl_xor(mx, 1));
                mx = fmaxf(mx, __shfl_xor(mx, 2));
                mx = fmaxf(mx, __shfl_xor(mx, 4));
                mx = fmaxf(mx, __shfl_xor(mx, 8));
                float nm    = fmaxf(mrow[rr], mx);
                float alpha = __expf(mrow[rr] - nm);
                float p0 = __expf(x0 - nm);
                float p1 = __expf(x1 - nm);
                float ps = p0 + p1;
                ps += __shfl_xor(ps, 1);
                ps += __shfl_xor(ps, 2);
                ps += __shfl_xor(ps, 4);
                ps += __shfl_xor(ps, 8);
                lrow[rr] = lrow[rr] * alpha + ps;
                mrow[rr] = nm;
#pragma unroll
                for (int nt = 0; nt < 8; ++nt) o[nt][rr] *= alpha;
                lP[w][rr + hi8][l15]      = (__bf16)p0;
                lP[w][rr + hi8][16 + l15] = (__bf16)p1;
            }
            asm volatile("s_wait_dscnt 0" ::: "memory");  // P store -> A-frag read (wave-local)

            bfrag ap;  // P as A-frag (16 x 32 over keys)
            ap.h2[0] = *(const v8bf*)&lP[w][l15][hi8];
            ap.h2[1] = *(const v8bf*)&lP[w][l15][16 + hi8];
#pragma unroll
            for (int nt = 0; nt < 8; ++nt) {
                bfrag bv;  // B[k=key][n=dk] = V[key][dk] = lVt[dk][key]
                bv.h2[0] = *(const v8bf*)&lVt[nt * 16 + l15][hi8];
                bv.h2[1] = *(const v8bf*)&lVt[nt * 16 + l15][16 + hi8];
                o[nt] = wmma_bf16(ap.v, bv.v, o[nt]);
            }
        }
        __syncthreads();
    }

    // normalize and store (heads concatenated along D -> matches reference reshape)
#pragma unroll
    for (int rr = 0; rr < 8; ++rr) {
        float inv = 1.0f / lrow[rr];
        int qi = q0 + rr + hi8;
        __bf16* orow = Op + (size_t)qi * D_MOD;
#pragma unroll
        for (int nt = 0; nt < 8; ++nt)
            orow[nt * 16 + l15] = (__bf16)(o[nt][rr] * inv);
    }
}

// ---------------------------------------------------------------------------
extern "C" void kernel_launch(void* const* d_in, const int* in_sizes, int n_in,
                              void* d_out, int out_size, void* d_ws, size_t ws_size,
                              hipStream_t stream) {
    (void)in_sizes; (void)n_in; (void)out_size; (void)ws_size;
    const float* x  = (const float*)d_in[0];
    const float* Wq = (const float*)d_in[1];
    const float* Wk = (const float*)d_in[2];
    const float* Wv = (const float*)d_in[3];
    const float* Wo = (const float*)d_in[4];
    float* out = (float*)d_out;

    char* ws = (char*)d_ws;
    size_t off = 0;
    auto alloc = [&](size_t elems) {
        __bf16* p = (__bf16*)(ws + off);
        off += elems * sizeof(__bf16);
        off = (off + 255) & ~(size_t)255;
        return p;
    };
    __bf16* xb  = alloc((size_t)M_ROWS * D_MOD);
    __bf16* Wqb = alloc((size_t)D_MOD * D_MOD);
    __bf16* Wkb = alloc((size_t)D_MOD * D_MOD);
    __bf16* Wvb = alloc((size_t)D_MOD * D_MOD);
    __bf16* Wob = alloc((size_t)D_MOD * D_MOD);
    __bf16* Qb  = alloc((size_t)M_ROWS * D_MOD);
    __bf16* Kb  = alloc((size_t)M_ROWS * D_MOD);
    __bf16* Vb  = alloc((size_t)M_ROWS * D_MOD);
    __bf16* Ob  = alloc((size_t)M_ROWS * D_MOD);

    const int nx = M_ROWS * D_MOD;   // 8388608
    const int nw = D_MOD * D_MOD;    // 4194304
    cvt_f32_to_bf16<<<nx / 1024, 256, 0, stream>>>(x,  xb,  nx);
    cvt_f32_to_bf16<<<nw / 1024, 256, 0, stream>>>(Wq, Wqb, nw);
    cvt_f32_to_bf16<<<nw / 1024, 256, 0, stream>>>(Wk, Wkb, nw);
    cvt_f32_to_bf16<<<nw / 1024, 256, 0, stream>>>(Wv, Wvb, nw);
    cvt_f32_to_bf16<<<nw / 1024, 256, 0, stream>>>(Wo, Wob, nw);

    dim3 gg(D_MOD / 128, M_ROWS / 128);   // (16, 32)
    gemm_bf16_nt<true><<<gg, 256, 0, stream>>>(xb, Wqb, Qb, M_ROWS, D_MOD, D_MOD);
    gemm_bf16_nt<true><<<gg, 256, 0, stream>>>(xb, Wkb, Kb, M_ROWS, D_MOD, D_MOD);
    gemm_bf16_nt<true><<<gg, 256, 0, stream>>>(xb, Wvb, Vb, M_ROWS, D_MOD, D_MOD);

    dim3 ga(S_LEN / 64, B_SZ * N_HEAD);   // (32, 32)
    flash_attn<<<ga, 128, 0, stream>>>(Qb, Kb, Vb, Ob);

    gemm_bf16_nt<false><<<gg, 256, 0, stream>>>(Ob, Wob, out, M_ROWS, D_MOD, D_MOD);
}
